// _MAB_23725399343797
// MI455X (gfx1250) — matrix operand order repaired and checked by reading, not compiled, not run on hardware
//
#include <hip/hip_runtime.h>
#include <hip/hip_bf16.h>
#include <math.h>

// ---------------------------------------------------------------------------
// MAB block (Set-Transformer style) for MI455X / gfx1250, wave32 + WMMA bf16
// + async global->LDS staging (ASYNCcnt) with double-buffered LDS tiles.
// Pipeline:
//   1) convert Q,K -> bf16 ; transpose+convert Wq/Wk/Wv/Wo -> bf16 [N][K]
//   2) WMMA GEMM projections: q (f32 + scaled bf16), k (bf16), v (bf16, transposed [B,H,dh,Nk])
//   3) flash attention per (b,h,q-tile): online softmax, P via LDS relayout, X1 = q + attn
//   4) LN1 -> O1 (f32 + bf16)
//   5) WMMA GEMM + exact-GELU residual epilogue: U = O1 + gelu(O1 @ Wo + bo)
//   6) LN2 -> out
// ---------------------------------------------------------------------------

#define USE_ASYNC 1   // async global->LDS copies (gfx1250); 0 = sync fallback

typedef unsigned short ushort_t;
typedef unsigned int   uint_t;
typedef __attribute__((ext_vector_type(16))) __bf16 v16bf;
typedef __attribute__((ext_vector_type(8)))  float  v8f;

struct Bf16x16 { uint4 lo, hi; };   // 32 bytes == 16 bf16

static __device__ __forceinline__ v8f v8f_zero() {
    v8f z = {0.f, 0.f, 0.f, 0.f, 0.f, 0.f, 0.f, 0.f};
    return z;
}

static __device__ __forceinline__ ushort_t f2bf(float x) {
    // round-to-nearest-even f32 -> bf16 (bit-level; avoids __bf16 arith)
    uint_t u = __builtin_bit_cast(uint_t, x);
    uint_t r = u + 0x7fffu + ((u >> 16) & 1u);
    return (ushort_t)(r >> 16);
}

static __device__ __forceinline__ v8f wmma_bf16(v16bf a, v16bf b, v8f c) {
    // (neg_a, A, neg_b, B, c_mod, C, reuse_a, reuse_b)
    return __builtin_amdgcn_wmma_f32_16x16x32_bf16(false, a, false, b, (short)0, c, false, false);
}

// Async 16B global -> LDS copy. VDST VGPR holds the LDS byte offset: for a
// flat pointer into shared memory, addr[31:0] IS the LDS offset (ISA 10.2
// aperture rules), so truncating the flat address gives the right operand.
static __device__ __forceinline__ void async_b128(const void* g, void* l) {
#if USE_ASYNC
    unsigned long long ga = (unsigned long long)(uintptr_t)g;
    unsigned lo = (unsigned)(uintptr_t)l;
    asm volatile("global_load_async_to_lds_b128 %0, %1, off"
                 :: "v"(lo), "v"(ga) : "memory");
#else
    *reinterpret_cast<uint4*>(l) = *reinterpret_cast<const uint4*>(g);
#endif
}

static __device__ __forceinline__ void wait_async() {
#if USE_ASYNC
    asm volatile("s_wait_asynccnt 0x0" ::: "memory");
#endif
}

// A-fragment (16x32, M x K): lane m = lane&15 ; K = 8*(lane>>4)+e (e<8) and 16+8*(lane>>4)+e (e>=8)
static __device__ __forceinline__ v16bf frag_a(const ushort_t* base, int pitch, int m0, int k0, int lane) {
    int m  = m0 + (lane & 15);
    int kb = k0 + ((lane >> 4) << 3);
    const ushort_t* p = base + (size_t)m * pitch + kb;
    Bf16x16 t;
    t.lo = *reinterpret_cast<const uint4*>(p);
    t.hi = *reinterpret_cast<const uint4*>(p + 16);
    return __builtin_bit_cast(v16bf, t);
}

// B-fragment (32x16, K x N) loaded from row-major [N][K] storage:
// lane n = lane&15 ; K = 8*(lane>>4)+e (e<8) and 16+8*(lane>>4)+e (e>=8)
static __device__ __forceinline__ v16bf frag_b(const ushort_t* base, int pitch, int n0, int k0, int lane) {
    int n  = n0 + (lane & 15);
    int kb = k0 + ((lane >> 4) << 3);
    const ushort_t* p = base + (size_t)n * pitch + kb;
    Bf16x16 t;
    t.lo = *reinterpret_cast<const uint4*>(p);
    t.hi = *reinterpret_cast<const uint4*>(p + 16);
    return __builtin_bit_cast(v16bf, t);
}

// ---------------------------------------------------------------------------
// Kernel 1a: f32 -> bf16 elementwise (vector-4)
// ---------------------------------------------------------------------------
__global__ __launch_bounds__(256) void k_cvt(const float* __restrict__ in,
                                             ushort_t* __restrict__ out, int n) {
    int i = (blockIdx.x * 256 + threadIdx.x) * 4;
    if (i + 3 < n) {
        float4 v = *reinterpret_cast<const float4*>(in + i);
        uint2 o;
        o.x = (uint_t)f2bf(v.x) | ((uint_t)f2bf(v.y) << 16);
        o.y = (uint_t)f2bf(v.z) | ((uint_t)f2bf(v.w) << 16);
        *reinterpret_cast<uint2*>(out + i) = o;
    }
}

// ---------------------------------------------------------------------------
// Kernel 1b: transpose+convert 512x512 weights: WT[n][k] = bf16(W[k][n])
// ---------------------------------------------------------------------------
__global__ __launch_bounds__(256) void k_wtrans(const float* __restrict__ w0,
                                                const float* __restrict__ w1,
                                                const float* __restrict__ w2,
                                                const float* __restrict__ w3,
                                                ushort_t* __restrict__ o0,
                                                ushort_t* __restrict__ o1,
                                                ushort_t* __restrict__ o2,
                                                ushort_t* __restrict__ o3) {
    const float* w = (blockIdx.y == 0) ? w0 : (blockIdx.y == 1) ? w1 : (blockIdx.y == 2) ? w2 : w3;
    ushort_t*    o = (blockIdx.y == 0) ? o0 : (blockIdx.y == 1) ? o1 : (blockIdx.y == 2) ? o2 : o3;
    int idx = blockIdx.x * 256 + threadIdx.x;   // 0 .. 512*512-1
    int n = idx >> 9, k = idx & 511;
    o[idx] = f2bf(w[k * 512 + n]);
}

// ---------------------------------------------------------------------------
// Kernel 2: bf16 WMMA GEMM, C = A[M x 512] * B[512 x 512] (+bias), tile 128x128.
// BT is B transposed [N][K] bf16. 256 threads = 8 waves (4 M-waves x 2 N-waves),
// each wave owns 32x64 (2x4 C fragments). K double-buffered through LDS in
// 64-chunks with async global->LDS copies overlapping the WMMA of the current
// chunk; s_wait_asynccnt 0 before each barrier.
// mode 0: projection epilogue (optional f32 / scaled-bf16 / transposed-bf16 outs)
// mode 1: U = res + gelu(acc + bias)   (exact GELU, erf)
// ---------------------------------------------------------------------------
#define GP 72   // LDS row pitch (64 + 8) -> conflict-free fragment reads

__global__ __launch_bounds__(256)
void k_gemm(const ushort_t* __restrict__ A, const ushort_t* __restrict__ BT,
            const float* __restrict__ bias,
            float* __restrict__ outF, ushort_t* __restrict__ outBF,
            ushort_t* __restrict__ outT, const float* __restrict__ res,
            float bfScale, int mode) {
    __shared__ __attribute__((aligned(16))) ushort_t As[2][128 * GP];
    __shared__ __attribute__((aligned(16))) ushort_t Bs[2][128 * GP];

    const int tid  = threadIdx.x;
    const int lane = tid & 31;
    const int w    = tid >> 5;
    const int mw   = w & 3;        // 0..3 -> M offset 32*mw
    const int nw   = w >> 2;       // 0..1 -> N offset 64*nw
    const int bm   = blockIdx.y * 128;
    const int bn   = blockIdx.x * 128;

    auto stage = [&](int buf, int kc) {
#pragma unroll
        for (int i = 0; i < 4; ++i) {
            int id = i * 256 + tid;
            int r = id >> 3, c = (id & 7) << 3;
            async_b128(A + (size_t)(bm + r) * 512 + kc + c, &As[buf][r * GP + c]);
            async_b128(BT + (size_t)(bn + r) * 512 + kc + c, &Bs[buf][r * GP + c]);
        }
    };

    v8f acc[2][4];
#pragma unroll
    for (int i = 0; i < 2; ++i)
#pragma unroll
        for (int j = 0; j < 4; ++j) acc[i][j] = v8f_zero();

    stage(0, 0);
    wait_async();
    __syncthreads();

    for (int kc = 0; kc < 512; kc += 64) {
        const int cur = (kc >> 6) & 1;
        if (kc + 64 < 512) stage(cur ^ 1, kc + 64);   // prefetch next chunk (async)

#pragma unroll
        for (int kk = 0; kk < 64; kk += 32) {
            v16bf a0 = frag_a(As[cur], GP, mw * 32,      kk, lane);
            v16bf a1 = frag_a(As[cur], GP, mw * 32 + 16, kk, lane);
#pragma unroll
            for (int j = 0; j < 4; ++j) {
                v16bf b = frag_b(Bs[cur], GP, nw * 64 + j * 16, kk, lane);
                acc[0][j] = wmma_bf16(a0, b, acc[0][j]);
                acc[1][j] = wmma_bf16(a1, b, acc[1][j]);
            }
        }
        wait_async();      // my copies into the other buffer are done
        __syncthreads();   // everyone done computing on cur + copies visible
    }

    const int hi = lane >> 4, lo = lane & 15;
#pragma unroll
    for (int i = 0; i < 2; ++i) {
#pragma unroll
        for (int j = 0; j < 4; ++j) {
            int n = bn + nw * 64 + j * 16 + lo;
            float bv = bias[n];
#pragma unroll
            for (int r = 0; r < 8; ++r) {
                int m = bm + mw * 32 + i * 16 + r + hi * 8;
                size_t idx = (size_t)m * 512 + n;
                float val = acc[i][j][r] + bv;
                if (mode == 0) {
                    if (outF)  outF[idx]  = val;
                    if (outBF) outBF[idx] = f2bf(val * bfScale);
                    if (outT) {
                        // v stored transposed: [B, H, dh, Nk]
                        int bb = m >> 10, key = m & 1023;
                        int h = n >> 6, d = n & 63;
                        outT[(((size_t)(bb * 8 + h) * 64 + d) << 10) + key] = f2bf(val);
                    }
                } else {
                    float g = 0.5f * val * (1.0f + erff(val * 0.70710678118f));
                    outF[idx] = res[idx] + g;
                }
            }
        }
    }
}

// ---------------------------------------------------------------------------
// Kernel 3: flash attention per (b, h, 64 q-rows). 128 threads = 4 waves,
// each wave owns 16 q rows. 128-key chunks double-buffered through LDS with
// async copies; online softmax in f32; P relaid C-layout -> A-layout via
// per-wave LDS tile; acc += P @ V^T. Writes X1 = q_f32 + attn
// (scores scale 1/sqrt(512) pre-folded into q_bf).
// ---------------------------------------------------------------------------
#define KP 72    // Ks pitch (64 + 8)
#define VP 136   // Vs pitch (128 + 8)
#define PP 136   // P  pitch (128 + 8)

__global__ __launch_bounds__(128)
void k_attn(const ushort_t* __restrict__ qbf, const ushort_t* __restrict__ kbf,
            const ushort_t* __restrict__ vT, const float* __restrict__ qf,
            float* __restrict__ X1) {
    __shared__ __attribute__((aligned(16))) ushort_t Ks[2][128 * KP];   // [key][d]
    __shared__ __attribute__((aligned(16))) ushort_t Vs[2][64 * VP];    // [d][key] (V^T)
    __shared__ __attribute__((aligned(16))) ushort_t Ps[4][16 * PP];    // per-wave P staging

    const int tid = threadIdx.x, lane = tid & 31, w = tid >> 5;
    const int b = blockIdx.z, h = blockIdx.y;
    const int q0 = blockIdx.x * 64 + w * 16;
    const int hi = lane >> 4, lo = lane & 15;

    const ushort_t* kg = kbf + (size_t)b * 1024 * 512 + h * 64;
    const ushort_t* vg = vT + (size_t)(b * 8 + h) * 64 * 1024;

    auto stage = [&](int buf, int kc) {
#pragma unroll
        for (int i = 0; i < 8; ++i) {   // Ks: 128 keys x 64 d (128B rows)
            int id = i * 128 + tid;
            int r = id >> 3, c = (id & 7) << 3;
            async_b128(kg + (size_t)(kc + r) * 512 + c, &Ks[buf][r * KP + c]);
        }
#pragma unroll
        for (int i = 0; i < 8; ++i) {   // Vs: 64 d x 128 keys (256B row slices)
            int id = i * 128 + tid;
            int r = id >> 4, c = (id & 15) << 3;
            async_b128(vg + (size_t)r * 1024 + kc + c, &Vs[buf][r * VP + c]);
        }
    };

    // resident q A-fragments (dh = 64 -> two 16x32 fragments); scale folded in.
    const ushort_t* qrow = qbf + ((size_t)(b * 1024 + q0) * 512 + h * 64);
    v16bf aq0 = frag_a(qrow, 512, 0, 0,  lane);
    v16bf aq1 = frag_a(qrow, 512, 0, 32, lane);

    v8f acc[4];
#pragma unroll
    for (int j = 0; j < 4; ++j) acc[j] = v8f_zero();
    float mrow[8], lrow[8];
#pragma unroll
    for (int r = 0; r < 8; ++r) { mrow[r] = -INFINITY; lrow[r] = 0.f; }

    ushort_t* myP = Ps[w];

    stage(0, 0);
    wait_async();
    __syncthreads();

    for (int kc = 0; kc < 1024; kc += 128) {
        const int cur = (kc >> 7) & 1;
        if (kc + 128 < 1024) stage(cur ^ 1, kc + 128);   // async prefetch next chunk

        // scores: 8 tiles of 16x16 over this 128-key chunk
        v8f s[8];
#pragma unroll
        for (int nt = 0; nt < 8; ++nt) {
            v16bf b0 = frag_b(Ks[cur], KP, nt * 16, 0,  lane);
            v16bf b1 = frag_b(Ks[cur], KP, nt * 16, 32, lane);
            v8f t = wmma_bf16(aq0, b0, v8f_zero());
            s[nt]  = wmma_bf16(aq1, b1, t);
        }

        // online softmax: row max across chunk, rescale running state
        float mnew[8], scl[8];
#pragma unroll
        for (int r = 0; r < 8; ++r) {
            float x = s[0][r];
#pragma unroll
            for (int nt = 1; nt < 8; ++nt) x = fmaxf(x, s[nt][r]);
            x = fmaxf(x, __shfl_xor(x, 1));
            x = fmaxf(x, __shfl_xor(x, 2));
            x = fmaxf(x, __shfl_xor(x, 4));
            x = fmaxf(x, __shfl_xor(x, 8));
            mnew[r] = fmaxf(mrow[r], x);
            scl[r]  = __expf(mrow[r] - mnew[r]);
            lrow[r] *= scl[r];
            mrow[r] = mnew[r];
        }
#pragma unroll
        for (int j = 0; j < 4; ++j)
#pragma unroll
            for (int r = 0; r < 8; ++r) acc[j][r] *= scl[r];

        // p = exp(s - m), accumulate row sums, stage P to LDS (C -> A relayout)
        float ps[8] = {0.f, 0.f, 0.f, 0.f, 0.f, 0.f, 0.f, 0.f};
#pragma unroll
        for (int nt = 0; nt < 8; ++nt) {
#pragma unroll
            for (int r = 0; r < 8; ++r) {
                float p = __expf(s[nt][r] - mnew[r]);
                ps[r] += p;
                myP[(r + hi * 8) * PP + nt * 16 + lo] = f2bf(p);
            }
        }
#pragma unroll
        for (int r = 0; r < 8; ++r) {
            float x = ps[r];
            x += __shfl_xor(x, 1);
            x += __shfl_xor(x, 2);
            x += __shfl_xor(x, 4);
            x += __shfl_xor(x, 8);
            lrow[r] += x;
        }
        __syncthreads();   // P visible for A-layout reads

        // acc += P(16x128) @ V^T tiles: key sub-chunks of 32
#pragma unroll
        for (int ns = 0; ns < 4; ++ns) {
            v16bf ap = frag_a(myP, PP, 0, ns * 32, lane);
#pragma unroll
            for (int j = 0; j < 4; ++j) {
                v16bf bv = frag_b(Vs[cur], VP, j * 16, ns * 32, lane);
                acc[j] = wmma_bf16(ap, bv, acc[j]);
            }
        }
        wait_async();      // next-chunk copies complete (this wave)
        __syncthreads();   // all waves done reading cur; buffers swappable
    }

    // X1 = q + attn  (attn = acc / l)
#pragma unroll
    for (int j = 0; j < 4; ++j) {
#pragma unroll
        for (int r = 0; r < 8; ++r) {
            int m = q0 + r + hi * 8;
            size_t idx = (size_t)(b * 1024 + m) * 512 + h * 64 + j * 16 + lo;
            X1[idx] = qf[idx] + acc[j][r] / lrow[r];
        }
    }
}

// ---------------------------------------------------------------------------
// Kernel 4: LayerNorm over last dim (512), no affine. One wave per row,
// 16 contiguous floats per lane. Optional bf16 copy of the output.
// ---------------------------------------------------------------------------
__global__ __launch_bounds__(256)
void k_ln(const float* __restrict__ in, float* __restrict__ outF,
          ushort_t* __restrict__ outBF) {
    const int lane = threadIdx.x & 31, w = threadIdx.x >> 5;
    const size_t row = (size_t)blockIdx.x * 8 + w;
    const float* p = in + row * 512 + lane * 16;

    float x[16];
    float s = 0.f, ss = 0.f;
#pragma unroll
    for (int i = 0; i < 4; ++i) {
        float4 v = *reinterpret_cast<const float4*>(p + i * 4);
        x[i * 4 + 0] = v.x; x[i * 4 + 1] = v.y; x[i * 4 + 2] = v.z; x[i * 4 + 3] = v.w;
        s  += v.x + v.y + v.z + v.w;
        ss += v.x * v.x + v.y * v.y + v.z * v.z + v.w * v.w;
    }
#pragma unroll
    for (int m = 1; m <= 16; m <<= 1) {
        s  += __shfl_xor(s,  m);
        ss += __shfl_xor(ss, m);
    }
    const float mu  = s * (1.0f / 512.0f);
    const float var = ss * (1.0f / 512.0f) - mu * mu;
    const float rs  = rsqrtf(var + 1e-5f);

    float* o = outF + row * 512 + lane * 16;
    float y[16];
#pragma unroll
    for (int i = 0; i < 16; ++i) y[i] = (x[i] - mu) * rs;
#pragma unroll
    for (int i = 0; i < 4; ++i) {
        float4 v; v.x = y[i * 4]; v.y = y[i * 4 + 1]; v.z = y[i * 4 + 2]; v.w = y[i * 4 + 3];
        *reinterpret_cast<float4*>(o + i * 4) = v;
    }
    if (outBF) {
        uint_t* ob = reinterpret_cast<uint_t*>(outBF + row * 512 + lane * 16);
#pragma unroll
        for (int i = 0; i < 8; ++i)
            ob[i] = (uint_t)f2bf(y[2 * i]) | ((uint_t)f2bf(y[2 * i + 1]) << 16);
    }
}

// ---------------------------------------------------------------------------
// Host launcher
// ---------------------------------------------------------------------------
extern "C" void kernel_launch(void* const* d_in, const int* in_sizes, int n_in,
                              void* d_out, int out_size, void* d_ws, size_t ws_size,
                              hipStream_t stream) {
    (void)in_sizes; (void)n_in; (void)out_size; (void)ws_size;

    const float* Q  = (const float*)d_in[0];
    const float* K  = (const float*)d_in[1];
    const float* Wq = (const float*)d_in[2];
    const float* bq = (const float*)d_in[3];
    const float* Wk = (const float*)d_in[4];
    const float* bk = (const float*)d_in[5];
    const float* Wv = (const float*)d_in[6];
    const float* bv = (const float*)d_in[7];
    const float* Wo = (const float*)d_in[8];
    const float* bo = (const float*)d_in[9];

    const size_t NROW  = 16384;           // B*N = 16*1024
    const size_t SZ_BF = NROW * 512 * 2;  // 16 MB
    const size_t SZ_F  = NROW * 512 * 4;  // 32 MB
    const size_t SZ_W  = 512 * 512 * 2;   // 512 KB

    char* ws = (char*)d_ws;
    size_t o = 0;
    ushort_t* Qbf  = (ushort_t*)(ws + o); o += SZ_BF;
    ushort_t* Kbf  = (ushort_t*)(ws + o); o += SZ_BF;
    ushort_t* WqT  = (ushort_t*)(ws + o); o += SZ_W;
    ushort_t* WkT  = (ushort_t*)(ws + o); o += SZ_W;
    ushort_t* WvT  = (ushort_t*)(ws + o); o += SZ_W;
    ushort_t* WoT  = (ushort_t*)(ws + o); o += SZ_W;
    ushort_t* qbfS = (ushort_t*)(ws + o); o += SZ_BF;   // q * 1/sqrt(512), bf16
    ushort_t* kbf  = (ushort_t*)(ws + o); o += SZ_BF;
    ushort_t* vTbf = (ushort_t*)(ws + o); o += SZ_BF;   // [B,H,dh,Nk]
    float*    qf   = (float*)(ws + o);    o += SZ_F;    // q projection f32
    float*    X1   = (float*)(ws + o);    o += SZ_F;    // q + attn
    float*    O1f  = (float*)(ws + o);    o += SZ_F;    // LN1 out f32
    ushort_t* O1bf = (ushort_t*)(ws + o); o += SZ_BF;   // LN1 out bf16
    float*    U    = (float*)(ws + o);    o += SZ_F;    // O1 + gelu(O1@Wo+bo)

    const int NEL = 16 * 1024 * 512;      // 8,388,608

    k_cvt<<<NEL / (256 * 4), 256, 0, stream>>>(Q, Qbf, NEL);
    k_cvt<<<NEL / (256 * 4), 256, 0, stream>>>(K, Kbf, NEL);
    k_wtrans<<<dim3(512 * 512 / 256, 4), 256, 0, stream>>>(Wq, Wk, Wv, Wo, WqT, WkT, WvT, WoT);

    const dim3 gg(512 / 128, NROW / 128);   // (4, 128)
    const float invSqrtD = 0.04419417382f;  // 1/sqrt(512)
    // q: f32 (residual) + scaled bf16 (attention A)
    k_gemm<<<gg, 256, 0, stream>>>(Qbf, WqT, bq, qf, qbfS, nullptr, nullptr, invSqrtD, 0);
    // k: bf16 [B,Nk,H,dh]
    k_gemm<<<gg, 256, 0, stream>>>(Kbf, WkT, bk, nullptr, kbf, nullptr, nullptr, 1.0f, 0);
    // v: bf16 transposed [B,H,dh,Nk]
    k_gemm<<<gg, 256, 0, stream>>>(Kbf, WvT, bv, nullptr, nullptr, vTbf, nullptr, 1.0f, 0);

    k_attn<<<dim3(1024 / 64, 8, 16), 128, 0, stream>>>(qbfS, kbf, vTbf, qf, X1);

    k_ln<<<NROW / 8, 256, 0, stream>>>(X1, O1f, O1bf);

    // U = O1 + gelu(O1 @ Wo + bo)
    k_gemm<<<gg, 256, 0, stream>>>(O1bf, WoT, bo, U, nullptr, nullptr, O1f, 1.0f, 1);

    k_ln<<<NROW / 8, 256, 0, stream>>>(U, (float*)d_out, nullptr);
}
